// LinearAttention_65025804862053
// MI455X (gfx1250) — compile-verified, matrix operand order checked by reading, not run on hardware
//
#include <hip/hip_runtime.h>
#include <hip/hip_bf16.h>

typedef __attribute__((ext_vector_type(16))) __bf16 v16bf;
typedef __attribute__((ext_vector_type(4)))  __bf16 v4bf;
typedef __attribute__((ext_vector_type(8)))  float  v8f;

constexpr int BATCH = 8;
constexpr int DIM   = 128;
constexpr int NTOK  = 128 * 256;   // 32768 tokens per batch
constexpr int CHUNK = 1024;        // tokens reduced per block in phase 1
constexpr int KSTEP = 32;          // bf16 WMMA K
constexpr int S1    = 36;          // LDS row stride (bf16) phase 1: 32 + 4 pad
constexpr int S2    = 132;         // LDS row stride (bf16) phase 2: 128 + 4 pad
constexpr float EPS = 1e-8f;

__device__ __forceinline__ float phi(float x) {
    // elu(x) + 1  ==  x>0 ? x+1 : exp(x)
    return x > 0.f ? x + 1.f : __expf(x);
}

__device__ __forceinline__ v4bf pack4(float a, float b, float c, float d) {
    v4bf p; p[0] = (__bf16)a; p[1] = (__bf16)b; p[2] = (__bf16)c; p[3] = (__bf16)d;
    return p;
}

// Load a 16x32 bf16 fragment (A or B operand) from LDS, rows striped per lane.
// Per CDNA5 §7.12.2: lane<16 holds K={0..7,16..23}; lane>=16 holds K={8..15,24..31}.
__device__ __forceinline__ v16bf ld_frag(const __bf16* lds, int row, int stride,
                                         int kbase, int lane) {
    const int koff = (lane < 16) ? 0 : 8;
    const __bf16* p = lds + row * stride + kbase + koff;
    v16bf f;
#pragma unroll
    for (int j = 0; j < 8; ++j) { f[j] = p[j]; f[j + 8] = p[j + 16]; }
    return f;
}

__global__ void la_zero(float* p, int n) {
    int i = blockIdx.x * blockDim.x + threadIdx.x;
    if (i < n) p[i] = 0.f;
}

// ---------------- Phase 1: KV state + K_sum (token reduction) ----------------
// grid = BATCH * (NTOK/CHUNK) blocks of 256 threads (8 waves).
// Wave w owns d-strip [16w,16w+16) x all 128 e columns (8 tiles of 16).
__global__ void __launch_bounds__(256)
la_phase1(const float* __restrict__ kin, const float* __restrict__ vin,
          float* __restrict__ kvout, float* __restrict__ ksumout) {
    __shared__ __bf16 sK[DIM * S1];   // phi(K): rows d, cols token-chunk
    __shared__ __bf16 sV[DIM * S1];   // V     : rows e, cols token-chunk
    __shared__ float  sKsum[DIM];

    const int blocksPerBatch = NTOK / CHUNK;
    const int b  = blockIdx.x / blocksPerBatch;
    const int n0 = (blockIdx.x % blocksPerBatch) * CHUNK;
    const float* kb = kin + (size_t)b * DIM * NTOK;
    const float* vb = vin + (size_t)b * DIM * NTOK;

    const int t = threadIdx.x, lane = t & 31, w = t >> 5, laneM = lane & 15;

    if (t < DIM) sKsum[t] = 0.f;

    const v8f vzero = {0.f, 0.f, 0.f, 0.f, 0.f, 0.f, 0.f, 0.f};
    v8f acc[8];
#pragma unroll
    for (int T = 0; T < 8; ++T) acc[T] = vzero;
    __syncthreads();

    for (int ks = 0; ks < CHUNK / KSTEP; ++ks) {
        const int nb = n0 + ks * KSTEP;

        // Prefetch the next 32-token chunk (one row per thread t<128).
        if (t < DIM && ks + 1 < CHUNK / KSTEP) {
            __builtin_prefetch(kb + (size_t)t * NTOK + nb + KSTEP, 0, 0);
            __builtin_prefetch(vb + (size_t)t * NTOK + nb + KSTEP, 0, 0);
        }

        // Stage 128x32 of phi(K) and V as bf16. float4 loads (coalesced),
        // loads issued before any convert/store so they can clause.
        float4 kx[4], vx[4];
#pragma unroll
        for (int i = 0; i < 4; ++i) {
            const int flat = t + i * 256;          // 1024 float4s total
            const int d = flat >> 3, nc = (flat & 7) * 4;
            kx[i] = *(const float4*)(kb + (size_t)d * NTOK + nb + nc);
            vx[i] = *(const float4*)(vb + (size_t)d * NTOK + nb + nc);
        }
#pragma unroll
        for (int i = 0; i < 4; ++i) {
            const int flat = t + i * 256;
            const int d = flat >> 3, nc = (flat & 7) * 4;
            const float p0 = phi(kx[i].x), p1 = phi(kx[i].y);
            const float p2 = phi(kx[i].z), p3 = phi(kx[i].w);
            *(v4bf*)(&sK[d * S1 + nc]) = pack4(p0, p1, p2, p3);   // ds_store_b64
            atomicAdd(&sKsum[d], (p0 + p1) + (p2 + p3));          // ds_add_f32
            *(v4bf*)(&sV[d * S1 + nc]) = pack4(vx[i].x, vx[i].y, vx[i].z, vx[i].w);
        }
        __syncthreads();

        v16bf a = ld_frag(sK, 16 * w + laneM, S1, 0, lane);
#pragma unroll
        for (int T = 0; T < 8; ++T) {
            v16bf bf = ld_frag(sV, 16 * T + laneM, S1, 0, lane);
            acc[T] = __builtin_amdgcn_wmma_f32_16x16x32_bf16(
                false, a, false, bf, (short)0, acc[T], false, false);
        }
        __syncthreads();
    }

    // Accumulate partial KV into workspace. C layout: VGPR r -> M = r (+8 for
    // upper half-wave), N = lane%16.
    float* kvb = kvout + (size_t)b * DIM * DIM;
    const int moff = (lane < 16) ? 0 : 8;
#pragma unroll
    for (int T = 0; T < 8; ++T) {
        const int e = 16 * T + laneM;
#pragma unroll
        for (int r = 0; r < 8; ++r) {
            const int d = 16 * w + moff + r;
            unsafeAtomicAdd(&kvb[(size_t)d * DIM + e], acc[T][r]);
        }
    }
    if (t < DIM) unsafeAtomicAdd(&ksumout[b * DIM + t], sKsum[t]);
}

// ---------------- Phase 2: V_new = (phi(Q) @ KV) * Z ----------------
// grid = BATCH * (NTOK/64) blocks of 256 threads. Block owns 64 tokens.
// 4 M-strips x 8 e-tiles; wave w -> M-strip w&3, e-tiles 4*(w>>2)..+3.
__global__ void __launch_bounds__(256)
la_phase2(const float* __restrict__ qin, const float* __restrict__ kvg,
          const float* __restrict__ ksumg, float* __restrict__ out) {
    __shared__ __bf16 sA[64 * S2];    // phi(Q) transposed: rows token, cols d
    __shared__ __bf16 sB[DIM * S2];   // KV transposed: rows e, cols d
    __shared__ float  sKsum[DIM];
    __shared__ float  sZ[64];

    const int tilesPerBatch = NTOK / 64;
    const int b  = blockIdx.x / tilesPerBatch;
    const int n0 = (blockIdx.x % tilesPerBatch) * 64;
    const float* qb  = qin + (size_t)b * DIM * NTOK;
    const float* kvb = kvg + (size_t)b * DIM * DIM;

    const int t = threadIdx.x, lane = t & 31, w = t >> 5, laneM = lane & 15;

    if (t < DIM) sKsum[t] = ksumg[b * DIM + t];

    // Stage KV (float4 loads, coalesced) -> bf16 LDS transposed [e][d].
#pragma unroll
    for (int i = 0; i < 16; ++i) {
        const int f4 = t + i * 256;                // 4096 float4s = 16384 elems
        const float4 x = ((const float4*)kvb)[f4];
        const int base = f4 * 4, d = base >> 7, e = base & 127;
        sB[(e + 0) * S2 + d] = (__bf16)x.x;
        sB[(e + 1) * S2 + d] = (__bf16)x.y;
        sB[(e + 2) * S2 + d] = (__bf16)x.z;
        sB[(e + 3) * S2 + d] = (__bf16)x.w;
    }
    // Stage phi(Q) (float4 loads along tokens) -> bf16 LDS transposed [n][d].
#pragma unroll
    for (int i = 0; i < 8; ++i) {
        const int f4 = t + i * 256;                // 2048 float4s = 8192 elems
        const int base = f4 * 4, d = base >> 6, nn = base & 63;
        const float4 x = *(const float4*)(qb + (size_t)d * NTOK + n0 + nn);
        sA[(nn + 0) * S2 + d] = (__bf16)phi(x.x);
        sA[(nn + 1) * S2 + d] = (__bf16)phi(x.y);
        sA[(nn + 2) * S2 + d] = (__bf16)phi(x.z);
        sA[(nn + 3) * S2 + d] = (__bf16)phi(x.w);
    }
    __syncthreads();

    // Z[n] = 1 / (phi(Q)[n,:] . K_sum + eps)
    if (t < 64) {
        float z = EPS;
        for (int d = 0; d < DIM; ++d) z += (float)sA[t * S2 + d] * sKsum[d];
        sZ[t] = 1.f / z;
    }
    __syncthreads();

    const int mstrip = w & 3;
    const int t0     = (w >> 2) * 4;
    const v8f vzero = {0.f, 0.f, 0.f, 0.f, 0.f, 0.f, 0.f, 0.f};
    v8f acc[4];
#pragma unroll
    for (int j = 0; j < 4; ++j) acc[j] = vzero;

#pragma unroll
    for (int ks = 0; ks < DIM / KSTEP; ++ks) {
        v16bf a = ld_frag(sA, 16 * mstrip + laneM, S2, ks * KSTEP, lane);
#pragma unroll
        for (int j = 0; j < 4; ++j) {
            v16bf bf = ld_frag(sB, 16 * (t0 + j) + laneM, S2, ks * KSTEP, lane);
            acc[j] = __builtin_amdgcn_wmma_f32_16x16x32_bf16(
                false, a, false, bf, (short)0, acc[j], false, false);
        }
    }

    // Scale by Z and store. For fixed lane, r=0..7 are consecutive tokens ->
    // two contiguous 16B runs per accumulator (global_store_b128).
    const int moff = (lane < 16) ? 0 : 8;
    float zv[8];
#pragma unroll
    for (int r = 0; r < 8; ++r) zv[r] = sZ[16 * mstrip + moff + r];

    float* ob = out + (size_t)b * DIM * NTOK;
#pragma unroll
    for (int j = 0; j < 4; ++j) {
        const int e = 16 * (t0 + j) + laneM;
        float* dst = ob + (size_t)e * NTOK + n0 + 16 * mstrip + moff;
        float4 lo = make_float4(acc[j][0] * zv[0], acc[j][1] * zv[1],
                                acc[j][2] * zv[2], acc[j][3] * zv[3]);
        float4 hi = make_float4(acc[j][4] * zv[4], acc[j][5] * zv[5],
                                acc[j][6] * zv[6], acc[j][7] * zv[7]);
        *(float4*)(dst)     = lo;
        *(float4*)(dst + 4) = hi;
    }
}

extern "C" void kernel_launch(void* const* d_in, const int* in_sizes, int n_in,
                              void* d_out, int out_size, void* d_ws, size_t ws_size,
                              hipStream_t stream) {
    const float* q = (const float*)d_in[0];
    const float* k = (const float*)d_in[1];
    const float* v = (const float*)d_in[2];
    float* out  = (float*)d_out;
    float* kv   = (float*)d_ws;                    // B*128*128 f32
    float* ksum = kv + (size_t)BATCH * DIM * DIM;  // B*128 f32

    const int zn = BATCH * DIM * DIM + BATCH * DIM;
    la_zero<<<(zn + 255) / 256, 256, 0, stream>>>(kv, zn);

    la_phase1<<<BATCH * (NTOK / CHUNK), 256, 0, stream>>>(k, v, kv, ksum);
    la_phase2<<<BATCH * (NTOK / 64),    256, 0, stream>>>(q, kv, ksum, out);
}
